// MaskedAttentionHead_57964878627274
// MI455X (gfx1250) — matrix operand order repaired and checked by reading, not compiled, not run on hardware
//
#include <hip/hip_runtime.h>

#define BB 4
#define SS 2048
#define EE 1024
#define HH 64

typedef __attribute__((ext_vector_type(16))) _Float16 v16h;
typedef __attribute__((ext_vector_type(8)))  float    v8f;
typedef __attribute__((ext_vector_type(8)))  unsigned v8u;

union Pack8h { _Float16 h[8]; float4 f; };

static __device__ __forceinline__ v8f wmma_f16(v16h a, v16h b, v8f c) {
  // D = A(16x32 f16) * B(32x16 f16) + C(16x16 f32)
  return __builtin_amdgcn_wmma_f32_16x16x32_f16(false, a, false, b, (short)0, c, false, false);
}

// Low 32 bits of a generic pointer to LDS == wave-relative LDS byte address.
static __device__ __forceinline__ unsigned lds_addr(const void* p) {
  return (unsigned)(uintptr_t)p;
}
static __device__ __forceinline__ void async_ld_b128(unsigned lds, const void* g) {
  asm volatile("global_load_async_to_lds_b128 %0, %1, off"
               :: "v"(lds), "v"((unsigned long long)(uintptr_t)g) : "memory");
}
static __device__ __forceinline__ void async_ld_b64(unsigned lds, const void* g) {
  asm volatile("global_load_async_to_lds_b64 %0, %1, off"
               :: "v"(lds), "v"((unsigned long long)(uintptr_t)g) : "memory");
}

// ---------------------------------------------------------------------------
// Kernel 0: pack Wq|Wk|Wv (fp32 [64,1024] each) -> Wh f16 [192,1024] row-major
// ---------------------------------------------------------------------------
__global__ __launch_bounds__(256) void wpack_k(const float* __restrict__ Wq,
                                               const float* __restrict__ Wk,
                                               const float* __restrict__ Wv,
                                               _Float16* __restrict__ Wh) {
  int idx = blockIdx.x * 256 + threadIdx.x;   // < 192*1024
  int n = idx >> 10, k = idx & 1023;
  const float* W = (n < 64) ? Wq : ((n < 128) ? Wk : Wv);
  Wh[idx] = (_Float16)W[(n & 63) * EE + k];
}

// ---------------------------------------------------------------------------
// Kernel 1: LayerNorm stats: one wave32 per token -> (mu, rstd)
// ---------------------------------------------------------------------------
__global__ __launch_bounds__(256) void lnstats_k(const float* __restrict__ x,
                                                 float* __restrict__ stats) {
  int token = blockIdx.x * 8 + (threadIdx.x >> 5);
  int lane  = threadIdx.x & 31;
  const float* row = x + (size_t)token * EE;
  float s = 0.f, ss = 0.f;
  #pragma unroll
  for (int i = 0; i < 8; ++i) {
    float4 v = *(const float4*)(row + lane * 4 + i * 128);
    s  += v.x + v.y + v.z + v.w;
    ss += v.x * v.x + v.y * v.y + v.z * v.z + v.w * v.w;
  }
  #pragma unroll
  for (int m = 16; m >= 1; m >>= 1) {
    s  += __shfl_xor(s,  m, 32);
    ss += __shfl_xor(ss, m, 32);
  }
  float mu   = s * (1.f / EE);
  float var  = ss * (1.f / EE) - mu * mu;
  float rstd = rsqrtf(var + 1e-5f);
  if (lane == 0) { stats[2 * token] = mu; stats[2 * token + 1] = rstd; }
}

// ---------------------------------------------------------------------------
// Kernel 2: fused LayerNorm-apply + QKV projection via WMMA.
// Block = 128 threads (4 waves), owns a 16-token M-tile.
// Outputs: Qh,Kh [B][S][64] f16 row-major; Vt [B][64][S] f16 (transposed).
// ---------------------------------------------------------------------------
__global__ __launch_bounds__(128) void qkv_k(const float* __restrict__ x,
                                             const float* __restrict__ gamma,
                                             const float* __restrict__ beta,
                                             const _Float16* __restrict__ Wh,
                                             const float* __restrict__ stats,
                                             _Float16* __restrict__ Qh,
                                             _Float16* __restrict__ Kh,
                                             _Float16* __restrict__ Vt) {
  __shared__ _Float16 aTile[16 * 36];   // 16 tokens x 32 k, padded stride 36

  const int tid  = threadIdx.x;
  const int wave = tid >> 5, lane = tid & 31;
  const int g = lane >> 4, r = lane & 15;
  const int tokenBase = blockIdx.x * 16;

  const int tTok = tid >> 3;
  const int tK   = (tid & 7) * 4;
  const size_t xBase = (size_t)(tokenBase + tTok) * EE;
  const float mu   = stats[2 * (tokenBase + tTok)];
  const float rstd = stats[2 * (tokenBase + tTok) + 1];

  v8f acc0 = {}, acc1 = {}, acc2 = {};

  for (int c = 0; c < 32; ++c) {
    const int kb = c * 32;
    float4 xv = *(const float4*)(x + xBase + kb + tK);
    float4 gv = *(const float4*)(gamma + kb + tK);
    float4 bv = *(const float4*)(beta  + kb + tK);
    _Float16* dst = &aTile[tTok * 36 + tK];
    dst[0] = (_Float16)((xv.x - mu) * rstd * gv.x + bv.x);
    dst[1] = (_Float16)((xv.y - mu) * rstd * gv.y + bv.y);
    dst[2] = (_Float16)((xv.z - mu) * rstd * gv.z + bv.z);
    dst[3] = (_Float16)((xv.w - mu) * rstd * gv.w + bv.w);
    __syncthreads();

    v8u au;
    #pragma unroll
    for (int i = 0; i < 8; ++i) {
      const int k0 = (i < 4) ? (2 * i + 8 * g) : (16 + 2 * (i - 4) + 8 * g);
      au[i] = *(const unsigned*)&aTile[r * 36 + k0];
    }
    const v16h af = __builtin_bit_cast(v16h, au);

    #pragma unroll
    for (int i = 0; i < 3; ++i) {
      const int nt = wave * 3 + i;                      // n-tile 0..11 (Q,K,V x 4)
      const _Float16* wrow = Wh + (size_t)(nt * 16 + r) * EE + kb;
      v8u bu;
      #pragma unroll
      for (int j = 0; j < 8; ++j)
        bu[j] = *(const unsigned*)&wrow[(j < 4) ? (2 * j + 8 * g) : (16 + 2 * (j - 4) + 8 * g)]; // table-order k-pair
      const v16h bf = __builtin_bit_cast(v16h, bu);
      v8f& acc = (i == 0) ? acc0 : ((i == 1) ? acc1 : acc2);
      acc = wmma_f16(af, bf, acc);
    }
    __syncthreads();
  }

  const int b = tokenBase / SS;
  const int sBase = tokenBase % SS;
  #pragma unroll
  for (int i = 0; i < 3; ++i) {
    const v8f& acc = (i == 0) ? acc0 : ((i == 1) ? acc1 : acc2);
    const int nt  = wave * 3 + i;
    const int mat = nt >> 2;                 // 0=Q, 1=K, 2=V
    const int ht  = nt & 3;
    if (mat < 2) {
      _Float16* dstM = (mat == 0 ? Qh : Kh) +
                       ((size_t)b * SS + sBase) * HH + ht * 16 + r;
      #pragma unroll
      for (int v = 0; v < 8; ++v)
        dstM[(size_t)(v + 8 * g) * HH] = (_Float16)acc[v];
    } else {
      Pack8h pk;
      #pragma unroll
      for (int v = 0; v < 8; ++v) pk.h[v] = (_Float16)acc[v];
      _Float16* dstV = Vt + ((size_t)b * HH + ht * 16 + r) * SS + sBase + 8 * g;
      *(float4*)dstV = pk.f;
    }
  }
}

// ---------------------------------------------------------------------------
// Kernel 3: causal flash attention.
// Block = 8 waves owning 8 consecutive 16-query tiles; K/V tiles (32 keys)
// are cooperatively staged in LDS with double-buffered async-to-LDS copies.
// ---------------------------------------------------------------------------
__global__ __launch_bounds__(256) void attn_k(const _Float16* __restrict__ Qh,
                                              const _Float16* __restrict__ Kh,
                                              const _Float16* __restrict__ Vt,
                                              float* __restrict__ out) {
  // K tile: 32 keys x 64 h, row stride 144B (pad) -> 16B-aligned b128 async writes
  __shared__ _Float16 kBuf[2][32 * 72];
  // V tile: 64 h x 32 keys, row stride 80B (pad) -> 8B-aligned b64 async writes
  __shared__ _Float16 vBuf[2][64 * 40];
  __shared__ _Float16 pTile[8][16 * 36];   // per-wave P transpose staging

  const int tid  = threadIdx.x;
  const int wave = tid >> 5, lane = tid & 31;
  const int g = lane >> 4, r = lane & 15;

  const int tile0 = blockIdx.x * 8;          // first query tile of block
  const int b   = tile0 >> 7;                // 128 tiles per batch
  const int qt0 = tile0 & 127;
  const int qBase = (qt0 + wave) * 16;       // this wave's query tile

  const _Float16* Qb = Qh + (size_t)b * SS * HH;
  const char*     KbB = (const char*)(Kh + (size_t)b * SS * HH);
  const char*     VbB = (const char*)(Vt + (size_t)b * HH * SS);
  _Float16* myP = &pTile[wave][0];

  // ---- cooperative async staging of one 32-key K/V tile (3 ops per thread)
  auto issueTile = [&](int it) {
    const int buf = it & 1;
    const int kBase = it * 32;
    {
      const int row = tid >> 3, seg = tid & 7;        // K: 32 rows x 8 x 16B
      async_ld_b128(lds_addr(&kBuf[buf][0]) + row * 144 + seg * 16,
                    KbB + (size_t)(kBase + row) * 128 + seg * 16);
    }
    #pragma unroll
    for (int half = 0; half < 2; ++half) {            // V: 64 rows x 8 x 8B
      const int cid = tid + half * 256;
      const int row = cid >> 3, seg = cid & 7;
      async_ld_b64(lds_addr(&vBuf[buf][0]) + row * 80 + seg * 8,
                   VbB + (size_t)row * 4096 + (size_t)kBase * 2 + seg * 8);
    }
  };

  // ---- Q A-fragments for both 32-wide h chunks, kept in registers
  v16h qf0, qf1;
  {
    v8u u0, u1;
    #pragma unroll
    for (int i = 0; i < 8; ++i) {
      const int k0 = (i < 4) ? (2 * i + 8 * g) : (16 + 2 * (i - 4) + 8 * g);
      const size_t rowOff = (size_t)(qBase + r) * HH;
      u0[i] = *(const unsigned*)&Qb[rowOff + k0];
      u1[i] = *(const unsigned*)&Qb[rowOff + 32 + k0];
    }
    qf0 = __builtin_bit_cast(v16h, u0);
    qf1 = __builtin_bit_cast(v16h, u1);
  }

  v8f o0 = {}, o1 = {}, o2 = {}, o3 = {};
  float mrun[8], lrun[8];
  #pragma unroll
  for (int v = 0; v < 8; ++v) { mrun[v] = -3.0e38f; lrun[v] = 0.f; }

  const int myEnd = qBase + 16;                       // keys this wave needs
  const int nIter = (qt0 * 16 + 128 + 31) >> 5;       // block-uniform trip count

  issueTile(0);
  for (int it = 0; it < nIter; ++it) {
    const int buf = it & 1;
    const int kBase = it * 32;

    if (it + 1 < nIter) {
      issueTile(it + 1);                              // prefetch next tile
      asm volatile("s_wait_asynccnt 0x3" ::: "memory"); // current tile's 3 done
    } else {
      asm volatile("s_wait_asynccnt 0x0" ::: "memory");
    }
    __syncthreads();                                  // staged tile visible

    if (kBase < myEnd) {
      const char* kT = (const char*)&kBuf[buf][0];
      const char* vT = (const char*)&vBuf[buf][0];

      // ---- scores S = Q K^T : 4 WMMA (2 key halves x 2 h chunks)
      v8f s0 = {}, s1 = {};
      {
        const char* kR0 = kT + r * 144;
        const char* kR1 = kT + (16 + r) * 144;
        v8u b00, b01, b10, b11;
        #pragma unroll
        for (int j = 0; j < 8; ++j) {
          const int off = (j < 4) ? (4 * j + 16 * g) : (32 + 4 * (j - 4) + 16 * g);
          b00[j] = *(const unsigned*)(kR0 + off);
          b01[j] = *(const unsigned*)(kR0 + 64 + off);
          b10[j] = *(const unsigned*)(kR1 + off);
          b11[j] = *(const unsigned*)(kR1 + 64 + off);
        }
        s0 = wmma_f16(qf0, __builtin_bit_cast(v16h, b00), s0);
        s0 = wmma_f16(qf1, __builtin_bit_cast(v16h, b01), s0);
        s1 = wmma_f16(qf0, __builtin_bit_cast(v16h, b10), s1);
        s1 = wmma_f16(qf1, __builtin_bit_cast(v16h, b11), s1);
      }

      // ---- causal mask + scale + online softmax (rows span 16-lane group)
      float alpha[8];
      #pragma unroll
      for (int v = 0; v < 8; ++v) {
        const int q  = qBase + v + 8 * g;
        const int k0 = kBase + r;
        const int k1 = kBase + 16 + r;
        float a0 = (k0 <= q) ? s0[v] * 0.125f : -3.0e38f;   // 1/sqrt(64)
        float a1 = (k1 <= q) ? s1[v] * 0.125f : -3.0e38f;
        float mx = fmaxf(a0, a1);
        #pragma unroll
        for (int msk = 1; msk < 16; msk <<= 1)
          mx = fmaxf(mx, __shfl_xor(mx, msk, 16));
        const float mnew = fmaxf(mrun[v], mx);
        const float al = __expf(mrun[v] - mnew);
        const float p0 = __expf(a0 - mnew);
        const float p1 = __expf(a1 - mnew);
        lrun[v] = lrun[v] * al + p0 + p1;     // lane-partial row sum
        mrun[v] = mnew;
        alpha[v] = al;
        myP[(v + 8 * g) * 36 + r]      = (_Float16)p0;   // D-layout -> LDS
        myP[(v + 8 * g) * 36 + 16 + r] = (_Float16)p1;
      }
      #pragma unroll
      for (int v = 0; v < 8; ++v) {
        o0[v] *= alpha[v]; o1[v] *= alpha[v];
        o2[v] *= alpha[v]; o3[v] *= alpha[v];
      }

      asm volatile("s_wait_dscnt 0x0" ::: "memory");  // P transpose ready

      // ---- A fragment of P (16 queries x 32 keys) from LDS
      v8u pu;
      #pragma unroll
      for (int i = 0; i < 8; ++i) {
        const int k0 = (i < 4) ? (2 * i + 8 * g) : (16 + 2 * (i - 4) + 8 * g);
        pu[i] = *(const unsigned*)&myP[r * 36 + k0];
      }
      const v16h pf = __builtin_bit_cast(v16h, pu);

      // ---- O += P V : 4 WMMA, B-fragments from staged V tile
      v8u vu0, vu1, vu2, vu3;
      #pragma unroll
      for (int j = 0; j < 8; ++j) {
        const int off = (j < 4) ? (4 * j + 16 * g) : (32 + 4 * (j - 4) + 16 * g);
        vu0[j] = *(const unsigned*)(vT + (r)      * 80 + off);
        vu1[j] = *(const unsigned*)(vT + (16 + r) * 80 + off);
        vu2[j] = *(const unsigned*)(vT + (32 + r) * 80 + off);
        vu3[j] = *(const unsigned*)(vT + (48 + r) * 80 + off);
      }
      o0 = wmma_f16(pf, __builtin_bit_cast(v16h, vu0), o0);
      o1 = wmma_f16(pf, __builtin_bit_cast(v16h, vu1), o1);
      o2 = wmma_f16(pf, __builtin_bit_cast(v16h, vu2), o2);
      o3 = wmma_f16(pf, __builtin_bit_cast(v16h, vu3), o3);

      asm volatile("s_wait_dscnt 0x0" ::: "memory");  // LDS reads drained
    }
    __syncthreads();          // safe to overwrite this buffer next time
  }

  // ---- finalize: reduce row sums across 16-lane group, normalize, store
  #pragma unroll
  for (int v = 0; v < 8; ++v) {
    float l = lrun[v];
    #pragma unroll
    for (int msk = 1; msk < 16; msk <<= 1) l += __shfl_xor(l, msk, 16);
    lrun[v] = 1.0f / l;
  }
  float* ob = out + ((size_t)b * SS + qBase) * HH;
  #pragma unroll
  for (int v = 0; v < 8; ++v) {
    const size_t rowOff = (size_t)(v + 8 * g) * HH;
    ob[rowOff + r]      = o0[v] * lrun[v];
    ob[rowOff + 16 + r] = o1[v] * lrun[v];
    ob[rowOff + 32 + r] = o2[v] * lrun[v];
    ob[rowOff + 48 + r] = o3[v] * lrun[v];
  }
}

// ---------------------------------------------------------------------------
extern "C" void kernel_launch(void* const* d_in, const int* in_sizes, int n_in,
                              void* d_out, int out_size, void* d_ws, size_t ws_size,
                              hipStream_t stream) {
  const float* x     = (const float*)d_in[0];
  const float* gamma = (const float*)d_in[1];
  const float* beta  = (const float*)d_in[2];
  const float* Wq    = (const float*)d_in[3];
  const float* Wk    = (const float*)d_in[4];
  const float* Wv    = (const float*)d_in[5];
  float* out = (float*)d_out;

  char* ws = (char*)d_ws;
  float*    stats = (float*)ws;                                   //  64 KB
  _Float16* Wh    = (_Float16*)(ws + 65536);                      // 384 KB
  _Float16* Qh    = (_Float16*)(ws + 65536 + 393216);             //   1 MB
  _Float16* Kh    = (_Float16*)(ws + 65536 + 393216 + 1048576);   //   1 MB
  _Float16* Vt    = (_Float16*)(ws + 65536 + 393216 + 2097152);   //   1 MB

  hipLaunchKernelGGL(wpack_k,   dim3(768),  dim3(256), 0, stream, Wq, Wk, Wv, Wh);
  hipLaunchKernelGGL(lnstats_k, dim3(1024), dim3(256), 0, stream, x, stats);
  hipLaunchKernelGGL(qkv_k,     dim3(512),  dim3(128), 0, stream,
                     x, gamma, beta, Wh, stats, Qh, Kh, Vt);
  hipLaunchKernelGGL(attn_k,    dim3(64),   dim3(256), 0, stream, Qh, Kh, Vt, out);
}
